// MultiHeadAttention_33492154974534
// MI455X (gfx1250) — matrix operand order repaired and checked by reading, not compiled, not run on hardware
//
#include <hip/hip_runtime.h>
#include <hip/hip_bf16.h>

typedef _Float16 f16;
typedef _Float16 v8h  __attribute__((ext_vector_type(8)));
typedef _Float16 v16h __attribute__((ext_vector_type(16)));
typedef float    v8f  __attribute__((ext_vector_type(8)));
typedef int      v4i  __attribute__((ext_vector_type(4)));

#define BB 4
#define TT 2048
#define CC 1024
#define HH 16
#define DD 64

#define WMMA_F16(A, B, C) \
  __builtin_amdgcn_wmma_f32_16x16x32_f16(false, (A), false, (B), (short)0, (C), false, false)

#if __has_builtin(__builtin_amdgcn_global_load_async_to_lds_b128)
#define HAVE_ASYNC_LDS 1
#else
#define HAVE_ASYNC_LDS 0
#endif

#if __has_builtin(__builtin_amdgcn_s_wait_asynccnt)
#define WAIT_ASYNC(n) __builtin_amdgcn_s_wait_asynccnt(n)
#else
#define WAIT_ASYNC(n) asm volatile("s_wait_asynccnt %0" ::"n"(n) : "memory")
#endif

// typed address-space casts via integer round trip (low 32 bits = LDS offset).
// b128 payload = int4; param types per hipcc diagnostic: AS1 v4i* (global src).
#define AS1_V4I(p) ((__attribute__((address_space(1))) v4i*)(uintptr_t)(p))
#define AS3_V4I(p) ((__attribute__((address_space(3))) v4i*)(uint32_t)(uintptr_t)(p))

static __device__ inline v16h cat16(v8h lo, v8h hi) {
  v16h r;
#pragma unroll
  for (int i = 0; i < 8; ++i) { r[i] = lo[i]; r[i + 8] = hi[i]; }
  return r;
}

// A-operand (16x32, f16) from row-major [M, ldk] at (row0, k0).
// Layout used here (contiguous k): lane L holds row L&15; lanes 0-15: K 0..15, lanes 16-31: K 16..31.
static __device__ inline v16h load_a16x32(const f16* __restrict__ src, int ldk,
                                          int row0, int k0, int lane) {
  const f16* p = src + (size_t)(row0 + (lane & 15)) * ldk + k0 + ((lane & 16) ? 16 : 0);
  v8h lo = *(const v8h*)(p);
  v8h hi = *(const v8h*)(p + 8);
  return cat16(lo, hi);
}

// B-operand (32x16, f16): B[k,n] = src[n0+n][k0+k] with src row-major [N, ldk].
// Layout used here (contiguous k): lane L holds column N = L&15; lanes 0-15: K 0..15, lanes 16-31: K 16..31.
static __device__ inline v16h load_b32x16(const f16* __restrict__ src, int ldk,
                                          int n0, int k0, int lane) {
  const f16* p = src + (size_t)(n0 + (lane & 15)) * ldk + k0 + ((lane & 16) ? 16 : 0);
  v8h lo = *(const v8h*)(p);
  v8h hi = *(const v8h*)(p + 8);
  return cat16(lo, hi);
}

// Same B-operand gather but from an LDS tile (row-major [N rows][ldk]).
static __device__ inline v16h lds_b32x16(const f16* tile, int ldk, int n0, int k0, int lane) {
  const f16* p = tile + (n0 + (lane & 15)) * ldk + k0 + ((lane & 16) ? 16 : 0);
  v8h lo = *(const v8h*)(p);
  v8h hi = *(const v8h*)(p + 8);
  return cat16(lo, hi);
}

// ---------------------------------------------------------------- prep kernels
__global__ void cast_f32_to_f16_kernel(const float* __restrict__ src,
                                       f16* __restrict__ dst, int n) {
  int i = blockIdx.x * blockDim.x + threadIdx.x;
  int stride = gridDim.x * blockDim.x;
  for (; i < n; i += stride) dst[i] = (f16)src[i];
}

// src fp32 row-major [K, N] -> dst f16 row-major [N, K]
__global__ void transpose_cast_kernel(const float* __restrict__ src,
                                      f16* __restrict__ dst, int N, int K) {
  int i = blockIdx.x * blockDim.x + threadIdx.x;
  int stride = gridDim.x * blockDim.x;
  int total = N * K;
  for (; i < total; i += stride) {
    int n = i / K;
    int k = i - n * K;
    dst[i] = (f16)src[(size_t)k * N + n];
  }
}

// ---------------------------------------------------------------- qkv GEMM
__global__ __launch_bounds__(128) void qkv_gemm_kernel(
    const f16* __restrict__ xh, const f16* __restrict__ wT,
    const float* __restrict__ bias,
    f16* __restrict__ q, f16* __restrict__ k, f16* __restrict__ vT) {
  const int lane = threadIdx.x & 31;
  const int wave = threadIdx.x >> 5;
  const int tile = blockIdx.x * 4 + wave;
  const int NT = (3 * CC) / 64;  // 48
  const int m0 = (tile / NT) * 16;
  const int n0 = (tile % NT) * 64;

  v8f acc[4] = {};
  for (int kk = 0; kk < CC; kk += 32) {
    __builtin_prefetch(xh + (size_t)m0 * CC + kk + 256, 0, 1);
    v16h a = load_a16x32(xh, CC, m0, kk, lane);
#pragma unroll
    for (int c = 0; c < 4; ++c) {
      v16h b = load_b32x16(wT, CC, n0 + c * 16, kk, lane);
      acc[c] = WMMA_F16(a, b, acc[c]);
    }
  }

  const int half = lane >> 4;
#pragma unroll
  for (int c = 0; c < 4; ++c) {
    int n = n0 + c * 16 + (lane & 15);
    float bv = bias[n];
    int which = n / CC;  // 0=q, 1=k, 2=v  (uniform per 16-lane half)
    int nc = n % CC;
    int h = nc / DD;
    int d = nc % DD;
#pragma unroll
    for (int r = 0; r < 8; ++r) {
      int m = m0 + r + 8 * half;
      int b = m / TT;
      int t = m % TT;
      f16 val = (f16)(acc[c][r] + bv);
      if (which == 0)
        q[(((size_t)(b * HH + h)) * TT + t) * DD + d] = val;
      else if (which == 1)
        k[(((size_t)(b * HH + h)) * TT + t) * DD + d] = val;
      else
        vT[(((size_t)(b * HH + h)) * DD + d) * TT + t] = val;
    }
  }
}

// ---------------------------------------------------------------- attention
// One workgroup (4 waves) shares (b,h); waves take 4 consecutive 16-query tiles.
// K (32x64) and V^T (64x32) blocks are staged to LDS cooperatively, double
// buffered with async global->LDS copies (ASYNCcnt) when available.
__global__ __launch_bounds__(128) void attn_kernel(
    const f16* __restrict__ q, const f16* __restrict__ kmat,
    const f16* __restrict__ vT, f16* __restrict__ yh) {
  __shared__ f16 kbuf[2][32][64];  // [buf][key][d]      8 KB
  __shared__ f16 vbuf[2][64][32];  // [buf][d][key]      8 KB
  __shared__ f16 plds[4][16][32];  // per-wave P staging 4 KB

  const int tid = threadIdx.x;
  const int lane = tid & 31;
  const int wave = tid >> 5;
  const int bh = blockIdx.x / (TT / 64);           // 0..63
  const int qbase = (blockIdx.x % (TT / 64)) * 64; // 64 queries per block
  const int b = bh / HH, h = bh % HH;
  const f16* qp = q + (size_t)bh * TT * DD;
  const f16* kp = kmat + (size_t)bh * TT * DD;
  const f16* vp = vT + (size_t)bh * DD * TT;
  const int q0 = qbase + wave * 16;
  const int half = lane >> 4;

  // cooperative stage of one 32-key block into buffer `buf`:
  // 256 chunks of 16B for K + 256 chunks for V, 128 threads x (2+2) copies.
  auto stage = [&](int buf, int kb) {
#pragma unroll
    for (int i = 0; i < 2; ++i) {
      int c = tid + i * 128;              // chunk id 0..255
      int kr = c >> 3, kc = (c & 7) * 8;  // K: row=key, col base (8 f16)
      const f16* gk = kp + (size_t)(kb + kr) * DD + kc;
      f16* lk = &kbuf[buf][kr][kc];
      int vr = c >> 2, vc = (c & 3) * 8;  // V: row=d, col base (8 f16)
      const f16* gv = vp + (size_t)vr * TT + kb + vc;
      f16* lv = &vbuf[buf][vr][vc];
#if HAVE_ASYNC_LDS
      __builtin_amdgcn_global_load_async_to_lds_b128(AS1_V4I(gk), AS3_V4I(lk), 0, 0);
      __builtin_amdgcn_global_load_async_to_lds_b128(AS1_V4I(gv), AS3_V4I(lv), 0, 0);
#else
      *(v8h*)lk = *(const v8h*)gk;
      *(v8h*)lv = *(const v8h*)gv;
#endif
    }
  };

  // Q A-operands, d = 0..31 and 32..63 (reused for every key block)
  v16h qa0 = load_a16x32(qp, DD, q0, 0, lane);
  v16h qa1 = load_a16x32(qp, DD, q0, 32, lane);

  v8f accv[4] = {};
  float mrow[8], lrow[8];
#pragma unroll
  for (int r = 0; r < 8; ++r) { mrow[r] = -__builtin_inff(); lrow[r] = 0.f; }

  stage(0, 0);  // prologue
  for (int kb = 0; kb < TT; kb += 32) {
    const int cur = (kb >> 5) & 1;
    if (kb + 32 < TT) {
      stage(cur ^ 1, kb + 32);   // prefetch next block
#if HAVE_ASYNC_LDS
      WAIT_ASYNC(4);             // 4 just issued may stay in flight; older done
#endif
    } else {
#if HAVE_ASYNC_LDS
      WAIT_ASYNC(0);
#endif
    }
    __syncthreads();             // current buffer visible to all 4 waves

    const f16* kt = &kbuf[cur][0][0];
    const f16* vt = &vbuf[cur][0][0];

    // S = (Q K^T) / sqrt(D) for 16 queries x 32 keys (two 16x16 tiles)
    v8f s[2] = {};
#pragma unroll
    for (int j = 0; j < 2; ++j) {
      v16h b0 = lds_b32x16(kt, 64, j * 16, 0, lane);
      s[j] = WMMA_F16(qa0, b0, s[j]);
      v16h b1 = lds_b32x16(kt, 64, j * 16, 32, lane);
      s[j] = WMMA_F16(qa1, b1, s[j]);
#pragma unroll
      for (int r = 0; r < 8; ++r) s[j][r] *= 0.125f;  // 1/sqrt(64)
    }

    // online softmax; row = r + 8*half lives across one 16-lane half
#pragma unroll
    for (int r = 0; r < 8; ++r) {
      float mx = fmaxf(s[0][r], s[1][r]);
      mx = fmaxf(mx, __shfl_xor(mx, 1, 32));
      mx = fmaxf(mx, __shfl_xor(mx, 2, 32));
      mx = fmaxf(mx, __shfl_xor(mx, 4, 32));
      mx = fmaxf(mx, __shfl_xor(mx, 8, 32));
      float mnew = fmaxf(mrow[r], mx);
      float scale = __expf(mrow[r] - mnew);
      float p0 = __expf(s[0][r] - mnew);
      float p1 = __expf(s[1][r] - mnew);
      s[0][r] = p0;
      s[1][r] = p1;
      float rs = p0 + p1;
      rs += __shfl_xor(rs, 1, 32);
      rs += __shfl_xor(rs, 2, 32);
      rs += __shfl_xor(rs, 4, 32);
      rs += __shfl_xor(rs, 8, 32);
      lrow[r] = lrow[r] * scale + rs;
      mrow[r] = mnew;
#pragma unroll
      for (int c = 0; c < 4; ++c) accv[c][r] *= scale;
    }

    // stage P (16x32 f16) row-major into this wave's LDS slice
#pragma unroll
    for (int j = 0; j < 2; ++j)
#pragma unroll
      for (int r = 0; r < 8; ++r)
        plds[wave][r + 8 * half][j * 16 + (lane & 15)] = (f16)s[j][r];
    asm volatile("s_wait_dscnt 0" ::: "memory");  // wave-local LDS RAW fence

    // reload P in A-operand layout
    v16h pa;
    {
      const f16* p = &plds[wave][lane & 15][(lane & 16) ? 16 : 0];
      v8h lo = *(const v8h*)p;
      v8h hi = *(const v8h*)(p + 8);
      pa = cat16(lo, hi);
    }

    // Y += P @ V : V operand rows are d (vbuf is [d][key], contiguous keys)
#pragma unroll
    for (int c = 0; c < 4; ++c) {
      v16h vb = lds_b32x16(vt, 32, c * 16, 0, lane);
      accv[c] = WMMA_F16(pa, vb, accv[c]);
    }

    __syncthreads();  // all waves done reading before buffer reuse
  }

  // normalize and store y as f16 [B, T, C] (row-major for the proj GEMM)
#pragma unroll
  for (int c = 0; c < 4; ++c) {
    int d = c * 16 + (lane & 15);
#pragma unroll
    for (int r = 0; r < 8; ++r) {
      int t = q0 + r + 8 * half;
      float invl = 1.0f / lrow[r];
      yh[((size_t)(b * TT + t)) * CC + h * DD + d] = (f16)(accv[c][r] * invl);
    }
  }
}

// ---------------------------------------------------------------- out proj
__global__ __launch_bounds__(128) void proj_gemm_kernel(
    const f16* __restrict__ yh, const f16* __restrict__ wT,
    const float* __restrict__ bias, float* __restrict__ out) {
  const int lane = threadIdx.x & 31;
  const int wave = threadIdx.x >> 5;
  const int tile = blockIdx.x * 4 + wave;
  const int NT = CC / 64;  // 16
  const int m0 = (tile / NT) * 16;
  const int n0 = (tile % NT) * 64;

  v8f acc[4] = {};
  for (int kk = 0; kk < CC; kk += 32) {
    __builtin_prefetch(yh + (size_t)m0 * CC + kk + 256, 0, 1);
    v16h a = load_a16x32(yh, CC, m0, kk, lane);
#pragma unroll
    for (int c = 0; c < 4; ++c) {
      v16h b = load_b32x16(wT, CC, n0 + c * 16, kk, lane);
      acc[c] = WMMA_F16(a, b, acc[c]);
    }
  }

  const int half = lane >> 4;
#pragma unroll
  for (int c = 0; c < 4; ++c) {
    int n = n0 + c * 16 + (lane & 15);
    float bv = bias[n];
#pragma unroll
    for (int r = 0; r < 8; ++r) {
      int m = m0 + r + 8 * half;
      out[(size_t)m * CC + n] = acc[c][r] + bv;
    }
  }
}

// ---------------------------------------------------------------- launch
extern "C" void kernel_launch(void* const* d_in, const int* in_sizes, int n_in,
                              void* d_out, int out_size, void* d_ws, size_t ws_size,
                              hipStream_t stream) {
  (void)in_sizes; (void)n_in; (void)out_size; (void)ws_size;
  const float* x      = (const float*)d_in[0];
  const float* w_attn = (const float*)d_in[1];
  const float* b_attn = (const float*)d_in[2];
  const float* w_proj = (const float*)d_in[3];
  const float* b_proj = (const float*)d_in[4];
  float* out = (float*)d_out;

  // workspace carve-out (~90 MB total)
  char* ws = (char*)d_ws;
  f16* xh     = (f16*)ws; ws += (size_t)BB * TT * CC * sizeof(f16);        // 16 MB
  f16* wattnT = (f16*)ws; ws += (size_t)3 * CC * CC * sizeof(f16);         //  6 MB
  f16* wprojT = (f16*)ws; ws += (size_t)CC * CC * sizeof(f16);             //  2 MB
  f16* q      = (f16*)ws; ws += (size_t)BB * HH * TT * DD * sizeof(f16);   // 16 MB
  f16* k      = (f16*)ws; ws += (size_t)BB * HH * TT * DD * sizeof(f16);   // 16 MB
  f16* vT     = (f16*)ws; ws += (size_t)BB * HH * DD * TT * sizeof(f16);   // 16 MB
  f16* yh     = (f16*)ws;                                                  // 16 MB

  cast_f32_to_f16_kernel<<<2048, 256, 0, stream>>>(x, xh, BB * TT * CC);
  transpose_cast_kernel<<<4096, 256, 0, stream>>>(w_attn, wattnT, 3 * CC, CC);
  transpose_cast_kernel<<<2048, 256, 0, stream>>>(w_proj, wprojT, CC, CC);
  qkv_gemm_kernel<<<6144, 128, 0, stream>>>(xh, wattnT, b_attn, q, k, vT);
  // 64 (b,h) * 32 query-chunks(64q) = 2048 blocks of 4 waves
  attn_kernel<<<2048, 128, 0, stream>>>(q, k, vT, yh);
  proj_gemm_kernel<<<2048, 128, 0, stream>>>(yh, wprojT, b_proj, out);
}
